// SwinTransformerBlock_67765993997104
// MI455X (gfx1250) — compile-verified, matrix operand order checked by reading, not run on hardware
//
#include <hip/hip_runtime.h>
#include <hip/hip_bf16.h>

// ---------------------------------------------------------------------------
// SwinV2 window attention block on gfx1250 (CDNA5), wave32 + WMMA f16.
// ---------------------------------------------------------------------------

typedef _Float16 v16h __attribute__((ext_vector_type(16)));
typedef _Float16 v8h  __attribute__((ext_vector_type(8)));
typedef float    v8f  __attribute__((ext_vector_type(8)));

#define NH   8
#define HD   32
#define NTOK 64
#define TWO_N 128
#define DIM  256
#define BW   512
#define NW   64
#define CPB  512

// workspace layout (bytes)
#define WS_T    ((size_t)0)                       // 225*8 f32
#define WS_BIAS ((size_t)8192)                    // 8*128*128 f32
#define WS_Q    ((size_t)532480)                  // 512*8*128*32 f16
#define WS_K    (WS_Q + (size_t)33554432)         // 512*8* 64*32 f16
#define WS_V    (WS_K + (size_t)16777216)         // 512*8*128*32 f16
#define WS_AO   (WS_V + (size_t)33554432)         // 512*128*256  f16

// Load a 16x32 f16 A-fragment (or 32x16 B-fragment) per ISA layout:
// lane<16 holds K = {0..7, 16..23}; lane>=16 holds K = {8..15, 24..31}.
// `seg` points at the 32-half contiguous K segment for this lane's row/col,
// and must be 16B aligned (row strides are multiples of 8 halves).
__device__ __forceinline__ v16h load_frag(const _Float16* seg, int sel) {
    v8h lo = *(const v8h*)(seg + sel * 8);
    v8h hi = *(const v8h*)(seg + 16 + sel * 8);
    v16h f;
#pragma unroll
    for (int i = 0; i < 8; ++i) { f[i] = lo[i]; f[i + 8] = hi[i]; }
    return f;
}

__device__ __forceinline__ v8f wmma_f16(v16h a, v16h b, v8f c) {
    return __builtin_amdgcn_wmma_f32_16x16x32_f16(false, a, false, b,
                                                  (short)0, c, false, false);
}

// ---------------------------------------------------------------------------
// 1) CPB MLP: 225 positions, 2 -> 512 (relu) -> 8
// ---------------------------------------------------------------------------
__global__ __launch_bounds__(256) void swin_cpb_mlp(
    const float* __restrict__ tbl, const float* __restrict__ w1,
    const float* __restrict__ b1, const float* __restrict__ w2,
    float* __restrict__ t_out)
{
    int p = threadIdx.x;
    if (p >= 225) return;
    float x0 = tbl[p * 2 + 0], x1 = tbl[p * 2 + 1];
    float acc[NH] = {};
    for (int j = 0; j < CPB; ++j) {
        float hsum = fmaxf(w1[j * 2] * x0 + w1[j * 2 + 1] * x1 + b1[j], 0.0f);
#pragma unroll
        for (int hh = 0; hh < NH; ++hh) acc[hh] += w2[hh * CPB + j] * hsum;
    }
#pragma unroll
    for (int hh = 0; hh < NH; ++hh) t_out[p * NH + hh] = acc[hh];
}

// ---------------------------------------------------------------------------
// 2) Bias build: rpb gather + 16*sigmoid + 2*amod quadrants -> (8,128,128)
// ---------------------------------------------------------------------------
__global__ __launch_bounds__(128) void swin_bias_build(
    const float* __restrict__ t_tab, const int* __restrict__ rel_index,
    const float* __restrict__ amod1, const float* __restrict__ amod2,
    float* __restrict__ biasb)
{
    int M = blockIdx.x, h = blockIdx.y, n = threadIdx.x;
    int m = M & 63, nn = n & 63;
    int idx = rel_index[m * 64 + nn];
    float r = t_tab[idx * NH + h];
    float rpb = 16.0f / (1.0f + __expf(-r));
    bool same_half = (M < 64) == (n < 64);
    const float* am = same_half ? amod1 : amod2;
    biasb[((size_t)h * TWO_N + M) * TWO_N + n] =
        rpb + 2.0f * am[((size_t)h * 64 + m) * 64 + nn];
}

// ---------------------------------------------------------------------------
// 3) QKV GEMM: [x1;x2] (65536x256) @ qkv_w^T (256x768) + bias, scatter q/k/v.
//    Block tile 64x64, 8 waves, wave tile 16x32 (2 WMMA per K-step).
// ---------------------------------------------------------------------------
__global__ __launch_bounds__(256) void swin_qkv_gemm(
    const float* __restrict__ x1, const float* __restrict__ x2,
    const float* __restrict__ qkv_w, const float* __restrict__ q_bias,
    const float* __restrict__ v_bias,
    _Float16* __restrict__ qws, _Float16* __restrict__ kws,
    _Float16* __restrict__ vws)
{
    __shared__ _Float16 sA[64 * 40];   // 64 rows x 32 halves, stride 40
    __shared__ _Float16 sB[64 * 40];

    const int tid   = threadIdx.x;
    const int baseM = blockIdx.y * 64;
    const int baseN = blockIdx.x * 64;
    const int wv = tid >> 5, lane = tid & 31;
    const int wr = wv & 3, wc = wv >> 2;
    const int sel = lane >> 4, lm = lane & 15;

    v8f acc0 = {}, acc1 = {};

    const int lrow  = tid >> 2;        // 0..63
    const int lpart = (tid & 3) * 8;   // 0,8,16,24
    const int grow  = baseM + lrow;
    const int which = grow >> 15;      // 0 -> x1, 1 -> x2
    const int loc   = grow & 32767;
    const float* xsrc = (which ? x2 : x1) + (size_t)loc * DIM;
    const float* wsrc = qkv_w + (size_t)(baseN + lrow) * DIM;

    for (int kk = 0; kk < DIM; kk += 32) {
        {
            const float* p = xsrc + kk + lpart;
            _Float16* d = &sA[lrow * 40 + lpart];
#pragma unroll
            for (int i = 0; i < 8; ++i) d[i] = (_Float16)p[i];
        }
        {
            const float* p = wsrc + kk + lpart;
            _Float16* d = &sB[lrow * 40 + lpart];
#pragma unroll
            for (int i = 0; i < 8; ++i) d[i] = (_Float16)p[i];
        }
        __syncthreads();
        v16h a  = load_frag(&sA[(wr * 16 + lm) * 40], sel);
        v16h b0 = load_frag(&sB[(wc * 32 + 0  + lm) * 40], sel);
        v16h b1 = load_frag(&sB[(wc * 32 + 16 + lm) * 40], sel);
        acc0 = wmma_f16(a, b0, acc0);
        acc1 = wmma_f16(a, b1, acc1);
        __syncthreads();
    }

#pragma unroll
    for (int nt = 0; nt < 2; ++nt) {
        const v8f& acc = nt ? acc1 : acc0;
        int col  = baseN + wc * 32 + nt * 16 + lm;
        int s    = col >> 8;           // 0=q,1=k,2=v
        int hcol = (col >> 5) & 7;
        int dd   = col & 31;
        float bv = (s == 0) ? q_bias[col] : ((s == 2) ? v_bias[col - 512] : 0.0f);
#pragma unroll
        for (int j = 0; j < 8; ++j) {
            int gr  = baseM + wr * 16 + j + sel * 8;
            int wch = gr >> 15;
            int lc  = gr & 32767;
            int b   = lc >> 6;
            int tok = lc & 63;
            _Float16 hv = (_Float16)(acc[j] + bv);
            if (s == 0) {
                qws[(((size_t)(b * NH + hcol)) * TWO_N + tok + wch * NTOK) * HD + dd] = hv;
            } else if (s == 1) {
                if (wch == 0)
                    kws[(((size_t)(b * NH + hcol)) * NTOK + tok) * HD + dd] = hv;
            } else {
                vws[(((size_t)(b * NH + hcol)) * TWO_N + tok + wch * NTOK) * HD + dd] = hv;
            }
        }
    }
}

// ---------------------------------------------------------------------------
// 4) Attention: per (b,h): S = (Qn Kn^T)*scale + bias + mask; softmax; O = P V
// ---------------------------------------------------------------------------
__global__ __launch_bounds__(256) void swin_attn(
    const _Float16* __restrict__ qws, const _Float16* __restrict__ kws,
    const _Float16* __restrict__ vws, const float* __restrict__ biasb,
    const float* __restrict__ mask, const float* __restrict__ logit_scale,
    _Float16* __restrict__ ao)
{
    __shared__ _Float16 sQ[TWO_N * 40];   // 128 x 32 (stride 40)
    __shared__ _Float16 sK[NTOK * 40];    //  64 x 32
    __shared__ _Float16 sVt[HD * 136];    //  32 x 128 (V transposed, stride 136)
    __shared__ _Float16 sP[TWO_N * 136];  // 128 x 128 probabilities, f16

    const int h = blockIdx.x, b = blockIdx.y;
    const int wdw = b & (NW - 1);         // window index for mask
    const int tid = threadIdx.x;
    const size_t qbase = (size_t)(b * NH + h) * TWO_N * HD;
    const size_t kbase = (size_t)(b * NH + h) * NTOK * HD;

    // --- stage + L2-normalize Q and K, transpose-stage V ---
    if (tid < TWO_N) {
        const _Float16* src = qws + qbase + (size_t)tid * HD;
        float tmp[HD]; float ss = 0.0f;
#pragma unroll
        for (int i = 0; i < HD; ++i) { tmp[i] = (float)src[i]; ss += tmp[i] * tmp[i]; }
        float inv = 1.0f / fmaxf(sqrtf(ss), 1e-12f);
#pragma unroll
        for (int i = 0; i < HD; ++i) sQ[tid * 40 + i] = (_Float16)(tmp[i] * inv);
    } else if (tid < TWO_N + NTOK) {
        int r = tid - TWO_N;
        const _Float16* src = kws + kbase + (size_t)r * HD;
        float tmp[HD]; float ss = 0.0f;
#pragma unroll
        for (int i = 0; i < HD; ++i) { tmp[i] = (float)src[i]; ss += tmp[i] * tmp[i]; }
        float inv = 1.0f / fmaxf(sqrtf(ss), 1e-12f);
#pragma unroll
        for (int i = 0; i < HD; ++i) sK[r * 40 + i] = (_Float16)(tmp[i] * inv);
    }
    if (tid < TWO_N) {
        const _Float16* src = vws + qbase + (size_t)tid * HD;
#pragma unroll
        for (int i = 0; i < HD; ++i) sVt[i * 136 + tid] = src[i];
    }
    __syncthreads();

    const float scale = __expf(fminf(logit_scale[h], 4.6051702f)); // log(100)

    const int wv = tid >> 5, lane = tid & 31;
    const int sel = lane >> 4, lm = lane & 15;

    // --- S = Q K^T : one WMMA per 16x16 tile (K-depth == head dim == 32) ---
    v16h aQ = load_frag(&sQ[(wv * 16 + lm) * 40], sel);
    v16h bK[4];
#pragma unroll
    for (int kb = 0; kb < 4; ++kb)
        bK[kb] = load_frag(&sK[(kb * 16 + lm) * 40], sel);

    v8f acc[8];
#pragma unroll
    for (int ct = 0; ct < 8; ++ct) {
        v8f z = {};
        acc[ct] = wmma_f16(aQ, bK[ct & 3], z);  // cols 64..127 reuse K (k=[k1;k1])
    }

    // --- scale + bias + mask, then row softmax (half-wave shfl reductions) ---
    const float* bptr = biasb + (size_t)h * TWO_N * TWO_N;
    const float* mptr = mask + (size_t)wdw * TWO_N * TWO_N;
#pragma unroll
    for (int j = 0; j < 8; ++j) {
        int M = wv * 16 + j + sel * 8;
        float mx = -1e30f;
#pragma unroll
        for (int ct = 0; ct < 8; ++ct) {
            int Nc = ct * 16 + lm;
            float v = acc[ct][j] * scale + bptr[M * TWO_N + Nc] + mptr[M * TWO_N + Nc];
            acc[ct][j] = v;
            mx = fmaxf(mx, v);
        }
        mx = fmaxf(mx, __shfl_xor(mx, 1, 32));
        mx = fmaxf(mx, __shfl_xor(mx, 2, 32));
        mx = fmaxf(mx, __shfl_xor(mx, 4, 32));
        mx = fmaxf(mx, __shfl_xor(mx, 8, 32));
        float sm = 0.0f;
#pragma unroll
        for (int ct = 0; ct < 8; ++ct) {
            float p = __expf(acc[ct][j] - mx);
            acc[ct][j] = p;
            sm += p;
        }
        sm += __shfl_xor(sm, 1, 32);
        sm += __shfl_xor(sm, 2, 32);
        sm += __shfl_xor(sm, 4, 32);
        sm += __shfl_xor(sm, 8, 32);
        float invs = 1.0f / sm;
#pragma unroll
        for (int ct = 0; ct < 8; ++ct) {
            int Nc = ct * 16 + lm;
            sP[M * 136 + Nc] = (_Float16)(acc[ct][j] * invs);
        }
    }
    __syncthreads();

    // --- O = P V : K-depth 128 -> 4 WMMA steps x 2 column tiles ---
    v8f o0 = {}, o1 = {};
#pragma unroll
    for (int kc = 0; kc < 4; ++kc) {
        v16h aP = load_frag(&sP[(wv * 16 + lm) * 136 + kc * 32], sel);
        v16h b0 = load_frag(&sVt[(0  + lm) * 136 + kc * 32], sel);
        v16h b1 = load_frag(&sVt[(16 + lm) * 136 + kc * 32], sel);
        o0 = wmma_f16(aP, b0, o0);
        o1 = wmma_f16(aP, b1, o1);
    }
#pragma unroll
    for (int nt = 0; nt < 2; ++nt) {
        const v8f& o = nt ? o1 : o0;
        int col = h * HD + nt * 16 + lm;
#pragma unroll
        for (int j = 0; j < 8; ++j) {
            int M = wv * 16 + j + sel * 8;
            ao[((size_t)b * TWO_N + M) * DIM + col] = (_Float16)o[j];
        }
    }
}

// ---------------------------------------------------------------------------
// 5) Output projection: ao (65536x256 f16) @ proj_w^T + proj_b -> f32 out
// ---------------------------------------------------------------------------
__global__ __launch_bounds__(256) void swin_proj_gemm(
    const _Float16* __restrict__ ao, const float* __restrict__ proj_w,
    const float* __restrict__ proj_b, float* __restrict__ out)
{
    __shared__ _Float16 sA[64 * 40];
    __shared__ _Float16 sB[64 * 40];

    const int tid   = threadIdx.x;
    const int baseM = blockIdx.y * 64;
    const int baseN = blockIdx.x * 64;
    const int wv = tid >> 5, lane = tid & 31;
    const int wr = wv & 3, wc = wv >> 2;
    const int sel = lane >> 4, lm = lane & 15;

    v8f acc0 = {}, acc1 = {};

    const int lrow  = tid >> 2;
    const int lpart = (tid & 3) * 8;
    const _Float16* asrc = ao + (size_t)(baseM + lrow) * DIM;
    const float*    wsrc = proj_w + (size_t)(baseN + lrow) * DIM;

    for (int kk = 0; kk < DIM; kk += 32) {
        *(v8h*)&sA[lrow * 40 + lpart] = *(const v8h*)(asrc + kk + lpart);
        {
            const float* p = wsrc + kk + lpart;
            _Float16* d = &sB[lrow * 40 + lpart];
#pragma unroll
            for (int i = 0; i < 8; ++i) d[i] = (_Float16)p[i];
        }
        __syncthreads();
        v16h a  = load_frag(&sA[(wr * 16 + lm) * 40], sel);
        v16h b0 = load_frag(&sB[(wc * 32 + 0  + lm) * 40], sel);
        v16h b1 = load_frag(&sB[(wc * 32 + 16 + lm) * 40], sel);
        acc0 = wmma_f16(a, b0, acc0);
        acc1 = wmma_f16(a, b1, acc1);
        __syncthreads();
    }

#pragma unroll
    for (int nt = 0; nt < 2; ++nt) {
        const v8f& acc = nt ? acc1 : acc0;
        int col = baseN + wc * 32 + nt * 16 + lm;
        float bv = proj_b[col];
#pragma unroll
        for (int j = 0; j < 8; ++j) {
            int gr = baseM + wr * 16 + j + sel * 8;
            out[(size_t)gr * DIM + col] = acc[j] + bv;
        }
    }
}

// ---------------------------------------------------------------------------
extern "C" void kernel_launch(void* const* d_in, const int* in_sizes, int n_in,
                              void* d_out, int out_size, void* d_ws, size_t ws_size,
                              hipStream_t stream)
{
    const float* x1          = (const float*)d_in[0];
    const float* x2          = (const float*)d_in[1];
    const float* mask        = (const float*)d_in[2];
    const float* qkv_w       = (const float*)d_in[3];
    const float* q_bias      = (const float*)d_in[4];
    const float* v_bias      = (const float*)d_in[5];
    const float* logit_scale = (const float*)d_in[6];
    const float* cpb_w1      = (const float*)d_in[7];
    const float* cpb_b1      = (const float*)d_in[8];
    const float* cpb_w2      = (const float*)d_in[9];
    const float* proj_w      = (const float*)d_in[10];
    const float* proj_b      = (const float*)d_in[11];
    const float* amod1       = (const float*)d_in[12];
    const float* amod2       = (const float*)d_in[13];
    const float* rel_table   = (const float*)d_in[14];
    const int*   rel_index   = (const int*)d_in[15];

    char* ws = (char*)d_ws;
    float*    t_tab = (float*)(ws + WS_T);
    float*    biasb = (float*)(ws + WS_BIAS);
    _Float16* qws   = (_Float16*)(ws + WS_Q);
    _Float16* kws   = (_Float16*)(ws + WS_K);
    _Float16* vws   = (_Float16*)(ws + WS_V);
    _Float16* ao    = (_Float16*)(ws + WS_AO);
    float*    out   = (float*)d_out;

    swin_cpb_mlp<<<1, 256, 0, stream>>>(rel_table, cpb_w1, cpb_b1, cpb_w2, t_tab);
    swin_bias_build<<<dim3(TWO_N, NH), 128, 0, stream>>>(t_tab, rel_index,
                                                         amod1, amod2, biasb);
    swin_qkv_gemm<<<dim3(12, 1024), 256, 0, stream>>>(x1, x2, qkv_w, q_bias,
                                                      v_bias, qws, kws, vws);
    swin_attn<<<dim3(NH, BW), 256, 0, stream>>>(qws, kws, vws, biasb, mask,
                                                logit_scale, ao);
    swin_proj_gemm<<<dim3(4, 1024), 256, 0, stream>>>(ao, proj_w, proj_b, out);
}
